// PhotonicCore_74758200754621
// MI455X (gfx1250) — matrix operand order repaired and checked by reading, not compiled, not run on hardware
//
#include <hip/hip_runtime.h>
#include <math.h>

typedef __attribute__((ext_vector_type(16))) _Float16 v16h;
typedef __attribute__((ext_vector_type(8)))  _Float16 v8h;
typedef __attribute__((ext_vector_type(8)))  float    v8f;

#define BDIM 256
#define NH   96          // harmonics
#define KCH  6           // k-chunks of 32 per j (192 features)
#define BT   64          // b rows per block
#define JT   32          // j per block (8 j-splits)
#define ICNT 256
#define BTOT 2048
#define OTOT 256

__device__ __forceinline__ unsigned pk16(float x, float y) {
  _Float16 a = (_Float16)x, b = (_Float16)y;
  unsigned short ua = __builtin_bit_cast(unsigned short, a);
  unsigned short ub = __builtin_bit_cast(unsigned short, b);
  return (unsigned)ua | ((unsigned)ub << 16);
}

// ---------------- Kernel 1: offset-side harmonic features (B operand) -------
// Layout (halves): (((j*6 + kc)*2 + s)*256 + i)*16 + e   -- each (j,kc) slice
// is a contiguous 16KB block already in WMMA-B lane order (s = K half).
__global__ void gen_g(const float* __restrict__ p, unsigned* __restrict__ g,
                      float sW, float sWq, float two_sq, float q) {
  const int j = blockIdx.x;
  const int i = threadIdx.x;
  float th = p[i * ICNT + j];
  float c1 = __cosf(th), s1 = __sinf(th);
  float up = sW, vp = 0.f;            // n=0 (weighted): +a0*cos(0), sin(0)
  float uc = sWq * c1, vc = sWq * s1; // n=1
  float beta = two_sq * c1;
#pragma unroll 4
  for (int n = 0; n < NH; ++n) {
    int k = 2 * n;
    int kc = k >> 5, kk = k & 31, s = kk >> 4, e = kk & 15;
    unsigned idx = ((((((unsigned)j * KCH + kc) * 2 + s) * 256 + i) * 16 + e) >> 1);
    g[idx] = pk16(uc, vc);
    float un = beta * uc - q * up;
    float vn = beta * vc - q * vp;
    up = uc; vp = vc; uc = un; vc = vn;
  }
}

// ---------------- Kernel 2: fused A-feature-gen + WMMA GEMM ------------------
__global__ void __launch_bounds__(BDIM)
photonic_wmma(const float* __restrict__ x, const unsigned* __restrict__ g,
              float* __restrict__ part,
              float phi0, float sW, float sWq, float two_sq, float q) {
  __shared__ _Float16 FxS[BT * 200];   // A features, stride 200 halves (16B aligned)
  __shared__ uint4    GlS[2][1024];    // double-buffered 16KB B chunk

  const int tid = threadIdx.x;
  const int lane = tid & 31, wave = tid >> 5;
  const int l16 = lane & 15, hi = lane >> 4;       // lane half selects K half
  const int msub = wave & 3, nhalf = wave >> 2;    // 4 M-subtiles x 2 N-halves
  const int bBase = blockIdx.x * BT;
  const int jBase = blockIdx.y * JT;

  v8f acc[8];
  for (int r = 0; r < 8; ++r)
    for (int e = 0; e < 8; ++e) acc[r][e] = 0.f;

  const uint4* g4 = (const uint4*)g;

  for (int jj = 0; jj < JT; ++jj) {
    const int j = jBase + jj;
    if (tid < BT) {
      // A-side features for 64 b-rows: weighted Chebyshev recurrence
      float th = phi0 + x[(size_t)(bBase + tid) * ICNT + j];
      float c1 = __cosf(th), s1 = __sinf(th);
      float up = -sW, vp = 0.f;              // cos side carries the minus sign
      float uc = -sWq * c1, vc = sWq * s1;
      float beta = two_sq * c1;
      const unsigned rb = (unsigned)tid * 200;
#pragma unroll 4
      for (int n = 0; n < NH; ++n) {
        *(unsigned*)&FxS[rb + 2 * n] = pk16(uc, vc);
        float un = beta * uc - q * up;
        float vn = beta * vc - q * vp;
        up = uc; vp = vc; uc = un; vc = vn;
      }
    } else {
      const uint4* src = g4 + ((size_t)j * KCH) * 1024;
      for (int ii = tid - BT; ii < 1024; ii += (BDIM - BT)) GlS[0][ii] = src[ii];
    }
    __syncthreads();

    for (int kc = 0; kc < KCH; ++kc) {
      const int cur = kc & 1;
      uint4 st0, st1, st2, st3;
      const bool pf = (kc + 1 < KCH);
      if (pf) { // software-pipelined prefetch of next B chunk (hides L2 latency)
        const uint4* src = g4 + ((size_t)j * KCH + kc + 1) * 1024;
        st0 = src[tid]; st1 = src[tid + 256]; st2 = src[tid + 512]; st3 = src[tid + 768];
      }
      // A fragment: lanes<16 -> K {0..15}, lanes>=16 -> K {16..31}
      const _Float16* fx = &FxS[(msub * 16 + l16) * 200 + kc * 32 + hi * 16];
      v16h a;
      *(v8h*)&a         = *(const v8h*)fx;
      *(((v8h*)&a) + 1) = *(const v8h*)(fx + 8);
      const _Float16* gl = (const _Float16*)GlS[cur];
#pragma unroll
      for (int r = 0; r < 8; ++r) {
        const int iloc = (nhalf * 8 + r) * 16 + l16;
        const _Float16* gp = gl + ((hi * 256 + iloc) * 16);
        v16h b;
        *(v8h*)&b         = *(const v8h*)gp;
        *(((v8h*)&b) + 1) = *(const v8h*)(gp + 8);
        acc[r] = __builtin_amdgcn_wmma_f32_16x16x32_f16(
            false, a, false, b, (short)0, acc[r], false, false);
      }
      if (pf) {
        GlS[1 - cur][tid]       = st0; GlS[1 - cur][tid + 256] = st1;
        GlS[1 - cur][tid + 512] = st2; GlS[1 - cur][tid + 768] = st3;
      }
      __syncthreads();
    }
  }

  // write deterministic per-j-split partials (no float atomics)
  float* pp = part + (size_t)blockIdx.y * ((size_t)BTOT * OTOT);
#pragma unroll
  for (int r = 0; r < 8; ++r) {
    const int i = (nhalf * 8 + r) * 16 + l16;
#pragma unroll
    for (int v = 0; v < 8; ++v) {
      const int b = bBase + msub * 16 + hi * 8 + v;
      pp[(size_t)b * OTOT + i] = acc[r][v];
    }
  }
}

// ---------------- Kernel 3: reduce partials + analytic base ------------------
__global__ void reduce_part(const float* __restrict__ part,
                            float* __restrict__ out, float basev) {
  const int idx = blockIdx.x * BDIM + threadIdx.x;
  float s = basev;
#pragma unroll
  for (int ps = 0; ps < 8; ++ps)
    s += part[(size_t)ps * ((size_t)BTOT * OTOT) + idx];
  out[idx] = s;
}

// ---------------- Fallback: direct transcendental kernel ---------------------
__global__ void photonic_direct(const float* __restrict__ x,
                                const float* __restrict__ p,
                                float* __restrict__ out,
                                float phi0, float u, float Dm, float Kc) {
  const int i = threadIdx.x;
  const int b = blockIdx.x;
  const float* xr = x + (size_t)b * ICNT;
  const float* pr = p + (size_t)i * ICNT;
  float s = 0.f;
  for (int j = 0; j < ICNT; ++j) {
    float ph = phi0 + xr[j] + pr[j];
    float c = __cosf(ph);
    float den = Dm - u * c;
    s += Kc / den;             // trans = 1 - Kc/den; fold the 1's into 256
  }
  out[(size_t)b * OTOT + i] = 256.0f - s;
}

extern "C" void kernel_launch(void* const* d_in, const int* in_sizes, int n_in,
                              void* d_out, int out_size, void* d_ws, size_t ws_size,
                              hipStream_t stream) {
  const float* x = (const float*)d_in[0];   // input_matrix [2048,256]
  const float* p = (const float*)d_in[1];   // phase_offset [256,256]
  float* out = (float*)d_out;

  const double kappa = 0.1, a = 0.99;
  const double t = sqrt(1.0 - kappa);
  const double q = a * t;                          // Poisson parameter = a*t
  const double Kc = kappa * (1.0 - a * a);         // (1-t^2)(1-a^2)
  const double W = 2.0 * Kc / (1.0 - q * q);
  const double R = 5e-6, neff = 3.48, lam = 1.55e-6;
  const double phi0 = fmod(2.0 * M_PI * neff * (2.0 * M_PI * R) / lam, 2.0 * M_PI);
  const float basev = (float)(256.0 * (1.0 - Kc / (1.0 - q * q)));

  const size_t needG = (size_t)256 * KCH * 2 * 256 * 16 * 2; // 25,165,824 B
  const size_t needP = (size_t)8 * BTOT * OTOT * 4;          // 16,777,216 B

  if (ws_size >= needG + needP) {
    unsigned* g = (unsigned*)d_ws;
    float* part = (float*)((char*)d_ws + needG);
    const float sW = (float)sqrt(W), sWq = (float)sqrt(W * q);
    const float tsq = (float)(2.0 * sqrt(q)), qf = (float)q;
    gen_g<<<256, BDIM, 0, stream>>>(p, g, sW, sWq, tsq, qf);
    photonic_wmma<<<dim3(BTOT / BT, OTOT / JT), BDIM, 0, stream>>>(
        x, g, part, (float)phi0, sW, sWq, tsq, qf);
    reduce_part<<<(BTOT * OTOT) / BDIM, BDIM, 0, stream>>>(part, out, basev);
  } else {
    photonic_direct<<<BTOT, BDIM, 0, stream>>>(
        x, p, out, (float)phi0, (float)(2.0 * a * t), (float)(1.0 + q * q),
        (float)Kc);
  }
}
